// GroupedQueryAttention_11836929868003
// MI455X (gfx1250) — compile-verified, matrix operand order checked by reading, not run on hardware
//
#include <hip/hip_runtime.h>
#include <hip/hip_bf16.h>

typedef __attribute__((ext_vector_type(16))) _Float16 v16h;
typedef __attribute__((ext_vector_type(8)))  _Float16 v8h;
typedef __attribute__((ext_vector_type(4)))  _Float16 v4h;
typedef __attribute__((ext_vector_type(8)))  float    v8f;

#define TSEQ     2048
#define EMB      2048
#define NHEADS   32
#define NKV      8
#define HDIM     64

// Load a 16-element WMMA fragment slice: elements 0-7 from p[0..7],
// elements 8-15 from p[16..23] (caller folds the +8*half lane offset into p).
// Both chunks are 16B-aligned -> two b128 loads (global or ds).
__device__ __forceinline__ v16h ldfrag(const _Float16* p) {
  v8h lo = *(const v8h*)p;
  v8h hi = *(const v8h*)(p + 16);
  return __builtin_shufflevector(lo, hi, 0, 1, 2, 3, 4, 5, 6, 7,
                                 8, 9, 10, 11, 12, 13, 14, 15);
}

// ---------------------------------------------------------------------------
// fp32 -> f16 bulk convert (4 elements/thread, bandwidth-bound).
// ---------------------------------------------------------------------------
__global__ __launch_bounds__(256)
void tohalf4(const float* __restrict__ in, _Float16* __restrict__ out) {
  int i = (blockIdx.x * 256 + threadIdx.x) * 4;
  float4 f = *(const float4*)(in + i);
  v4h h;
  h[0] = (_Float16)f.x; h[1] = (_Float16)f.y;
  h[2] = (_Float16)f.z; h[3] = (_Float16)f.w;
  *(v4h*)(out + i) = h;
}

// ---------------------------------------------------------------------------
// C[M x N] = A[M x K] * B[N x K]^T, A/B f16 row-major, C f32 row-major.
// Block = 128 threads = 4 waves stacked in M; each wave computes 32(M) x 64(N)
// -> 8 v_wmma_f32_16x16x32_f16 per 32-wide K step fed by 12 b128 loads.
// ---------------------------------------------------------------------------
__global__ __launch_bounds__(128)
void gemm_h(const _Float16* __restrict__ A, const _Float16* __restrict__ Bw,
            float* __restrict__ C, int Ndim, int Kdim) {
  const int wave = threadIdx.x >> 5;
  const int lane = threadIdx.x & 31;
  const int half = lane >> 4;
  const int l16  = lane & 15;
  const int dof  = 8 * half;

  const int mbase = (blockIdx.x * 4 + wave) * 32;
  const int nbase = blockIdx.y * 64;

  const _Float16* a0p = A + (size_t)(mbase + l16) * Kdim + dof;
  const _Float16* a1p = A + (size_t)(mbase + 16 + l16) * Kdim + dof;

  v8f acc[8] = {};

  for (int k0 = 0; k0 < Kdim; k0 += 32) {
    __builtin_prefetch(a0p + k0 + 64, 0, 0);
    v16h a0 = ldfrag(a0p + k0);
    v16h a1 = ldfrag(a1p + k0);
#pragma unroll
    for (int j = 0; j < 4; ++j) {
      const _Float16* bp = Bw + (size_t)(nbase + j * 16 + l16) * Kdim + k0 + dof;
      v16h b = ldfrag(bp);
      acc[j] = __builtin_amdgcn_wmma_f32_16x16x32_f16(
          false, a0, false, b, (short)0, acc[j], false, false);
      acc[4 + j] = __builtin_amdgcn_wmma_f32_16x16x32_f16(
          false, a1, false, b, (short)0, acc[4 + j], false, false);
    }
  }

  // C/D layout: VGPR r -> row r + 8*half, col = lane%16
#pragma unroll
  for (int j = 0; j < 4; ++j)
#pragma unroll
    for (int r = 0; r < 8; ++r) {
      int col = nbase + j * 16 + l16;
      C[(size_t)(mbase + r + 8 * half) * Ndim + col]      = acc[j][r];
      C[(size_t)(mbase + 16 + r + 8 * half) * Ndim + col] = acc[4 + j][r];
    }
}

// ---------------------------------------------------------------------------
// RoPE + fp32 -> f16 (scale folds 1/sqrt(HDIM) into Q). Layout [T][nheads][64].
// ---------------------------------------------------------------------------
__global__ __launch_bounds__(256)
void rope_f16(const float* __restrict__ in, _Float16* __restrict__ out,
              const float* __restrict__ cs, const float* __restrict__ sn,
              int nheads, float scale) {
  int idx = blockIdx.x * blockDim.x + threadIdx.x;
  int d = idx & 63;
  int t = idx / (nheads * 64);
  float v   = in[idx];
  float rot = (d < 32) ? -in[idx + 32] : in[idx - 32];
  float c = cs[t * 64 + d];
  float s = sn[t * 64 + d];
  out[idx] = (_Float16)((v * c + rot * s) * scale);
}

// ---------------------------------------------------------------------------
// V transpose to f16 [NKV*HDIM][T] so PV B-fragments are contiguous in key.
// ---------------------------------------------------------------------------
__global__ __launch_bounds__(256)
void vtrans(const float* __restrict__ V, _Float16* __restrict__ Vt) {
  int idx = blockIdx.x * blockDim.x + threadIdx.x;  // t*512 + c
  int c = idx & 511;
  int t = idx >> 9;
  Vt[(size_t)c * TSEQ + t] = (_Float16)V[idx];
}

// ---------------------------------------------------------------------------
// Flash-style causal attention. Grid = (T/64, NHEADS), block = 128 (4 waves).
// Each wave owns 16 query rows; key loop in steps of 32.
// Qh: f16 [T][32][64] (pre-scaled by 0.125), Kh: f16 [T][8][64],
// Vt: f16 [512][T], ctx out: f16 [T][2048] with col = h*64 + d.
// ---------------------------------------------------------------------------
__global__ __launch_bounds__(128)
void attn(const _Float16* __restrict__ Qh, const _Float16* __restrict__ Kh,
          const _Float16* __restrict__ Vt, _Float16* __restrict__ ctx) {
  __shared__ _Float16 plds[4][16 * 32];   // wave-private P tiles

  const int wave = threadIdx.x >> 5;
  const int lane = threadIdx.x & 31;
  const int half = lane >> 4;
  const int l16  = lane & 15;
  const int dof  = 8 * half;

  const int h   = blockIdx.y;
  const int hkv = h >> 2;                 // N_REP = 4
  const int qb  = blockIdx.x;             // 64 queries per block
  const int qwave = qb * 64 + wave * 16;

  // Q A-fragments for d 0..31 and 32..63 (row = l16)
  const _Float16* qp = Qh + ((size_t)(qwave + l16) * NHEADS + h) * HDIM + dof;
  v16h qa0 = ldfrag(qp);
  v16h qa1 = ldfrag(qp + 32);

  float m[8], l[8];
  v8f o[4] = {};
#pragma unroll
  for (int r = 0; r < 8; ++r) { m[r] = -1e30f; l[r] = 0.0f; }

  const int nsteps = 2 * qb + 2;          // keys 0 .. qb*64+63
  _Float16* pw = plds[wave];

  for (int s = 0; s < nsteps; ++s) {
    const int kbase = s * 32;

    // ---- S = Q @ K^T (two 16-key groups, two d-halves each) ----
    const _Float16* kp0 = Kh + ((size_t)(kbase + l16) * NKV + hkv) * HDIM + dof;
    const _Float16* kp1 = Kh + ((size_t)(kbase + 16 + l16) * NKV + hkv) * HDIM + dof;
    v8f s0 = {}, s1 = {};
    s0 = __builtin_amdgcn_wmma_f32_16x16x32_f16(false, qa0, false, ldfrag(kp0),
                                                (short)0, s0, false, false);
    s0 = __builtin_amdgcn_wmma_f32_16x16x32_f16(false, qa1, false, ldfrag(kp0 + 32),
                                                (short)0, s0, false, false);
    s1 = __builtin_amdgcn_wmma_f32_16x16x32_f16(false, qa0, false, ldfrag(kp1),
                                                (short)0, s1, false, false);
    s1 = __builtin_amdgcn_wmma_f32_16x16x32_f16(false, qa1, false, ldfrag(kp1 + 32),
                                                (short)0, s1, false, false);

    // ---- causal mask + online softmax (rows split by lane half) ----
    float alpha[8];
#pragma unroll
    for (int r = 0; r < 8; ++r) {
      const int q = qwave + r + 8 * half;
      float x0 = (kbase + l16      <= q) ? s0[r] : -1e30f;
      float x1 = (kbase + 16 + l16 <= q) ? s1[r] : -1e30f;
      float mx = fmaxf(x0, x1);
#pragma unroll
      for (int sh = 1; sh < 16; sh <<= 1) mx = fmaxf(mx, __shfl_xor(mx, sh, 32));
      float mn = fmaxf(m[r], mx);
      float a  = __expf(m[r] - mn);
      alpha[r] = a;
      m[r] = mn;
      float p0 = __expf(x0 - mn);
      float p1 = __expf(x1 - mn);
      float rs = p0 + p1;
#pragma unroll
      for (int sh = 1; sh < 16; sh <<= 1) rs += __shfl_xor(rs, sh, 32);
      l[r] = l[r] * a + rs;
      s0[r] = p0;
      s1[r] = p1;
    }

    // ---- transpose P (C-layout) -> A-fragment via wave-private LDS ----
#pragma unroll
    for (int r = 0; r < 8; ++r) {
      int row = r + 8 * half;
      pw[row * 32 + l16]      = (_Float16)s0[r];
      pw[row * 32 + 16 + l16] = (_Float16)s1[r];
    }
    __builtin_amdgcn_wave_barrier();
    asm volatile("s_wait_dscnt 0" ::: "memory");
    v16h pa = ldfrag(pw + l16 * 32 + dof);
    __builtin_amdgcn_wave_barrier();

    // ---- O = rescale(O) + P @ V ----
#pragma unroll
    for (int j = 0; j < 4; ++j) {
      const _Float16* vp =
          Vt + (size_t)(hkv * HDIM + j * 16 + l16) * TSEQ + kbase + dof;
      v16h vb = ldfrag(vp);
      v8f oc;
#pragma unroll
      for (int r = 0; r < 8; ++r) oc[r] = o[j][r] * alpha[r];
      o[j] = __builtin_amdgcn_wmma_f32_16x16x32_f16(
          false, pa, false, vb, (short)0, oc, false, false);
    }
  }

  // ---- epilogue: ctx[t][h*64+d] = O / l  (f16 for the Wo GEMM) ----
#pragma unroll
  for (int j = 0; j < 4; ++j)
#pragma unroll
    for (int r = 0; r < 8; ++r) {
      int row = qwave + r + 8 * half;
      int col = h * HDIM + j * 16 + l16;
      ctx[(size_t)row * EMB + col] = (_Float16)(o[j][r] / l[r]);
    }
}

// ---------------------------------------------------------------------------
extern "C" void kernel_launch(void* const* d_in, const int* in_sizes, int n_in,
                              void* d_out, int out_size, void* d_ws, size_t ws_size,
                              hipStream_t stream) {
  const float* x    = (const float*)d_in[0];
  const float* cosb = (const float*)d_in[1];
  const float* sinb = (const float*)d_in[2];
  const float* Wq   = (const float*)d_in[3];
  const float* Wk   = (const float*)d_in[4];
  const float* Wv   = (const float*)d_in[5];
  const float* Wo   = (const float*)d_in[6];
  float* out = (float*)d_out;

  char* ws = (char*)d_ws;
  float*    Qf   = (float*)(ws + ((size_t)0 << 20));     // 16 MB f32 Q proj
  float*    Kf   = (float*)(ws + ((size_t)16 << 20));    //  4 MB f32 K proj
  float*    Vf   = (float*)(ws + ((size_t)20 << 20));    //  4 MB f32 V proj
  _Float16* Qh   = (_Float16*)(ws + ((size_t)24 << 20)); //  8 MB f16 Q+rope
  _Float16* Kh   = (_Float16*)(ws + ((size_t)32 << 20)); //  2 MB f16 K+rope
  _Float16* Vt   = (_Float16*)(ws + ((size_t)34 << 20)); //  2 MB f16 V^T
  _Float16* ctxh = (_Float16*)(ws + ((size_t)36 << 20)); //  8 MB f16 ctx
  _Float16* xh   = (_Float16*)(ws + ((size_t)44 << 20)); //  8 MB f16 x
  _Float16* Wqh  = (_Float16*)(ws + ((size_t)52 << 20)); //  8 MB
  _Float16* Wkh  = (_Float16*)(ws + ((size_t)60 << 20)); //  2 MB
  _Float16* Wvh  = (_Float16*)(ws + ((size_t)62 << 20)); //  2 MB
  _Float16* Woh  = (_Float16*)(ws + ((size_t)64 << 20)); //  8 MB

  // --- bulk f32 -> f16 conversions (bandwidth-trivial) ---
  tohalf4<<<(TSEQ * EMB) / 1024, 256, 0, stream>>>(x, xh);
  tohalf4<<<(EMB * EMB) / 1024, 256, 0, stream>>>(Wq, Wqh);
  tohalf4<<<(NKV * HDIM * EMB) / 1024, 256, 0, stream>>>(Wk, Wkh);
  tohalf4<<<(NKV * HDIM * EMB) / 1024, 256, 0, stream>>>(Wv, Wvh);
  tohalf4<<<(EMB * EMB) / 1024, 256, 0, stream>>>(Wo, Woh);

  dim3 blk(128);
  // --- projections (pure-f16 WMMA GEMM, f32 out) ---
  gemm_h<<<dim3(TSEQ / 128, EMB / 64), blk, 0, stream>>>(xh, Wqh, Qf, EMB, EMB);
  gemm_h<<<dim3(TSEQ / 128, (NKV * HDIM) / 64), blk, 0, stream>>>(xh, Wkh, Kf, NKV * HDIM, EMB);
  gemm_h<<<dim3(TSEQ / 128, (NKV * HDIM) / 64), blk, 0, stream>>>(xh, Wvh, Vf, NKV * HDIM, EMB);

  // --- RoPE (+ fold 1/sqrt(64) into Q), f16 conversion ---
  rope_f16<<<(TSEQ * EMB) / 256, 256, 0, stream>>>(Qf, Qh, cosb, sinb, NHEADS, 0.125f);
  rope_f16<<<(TSEQ * NKV * HDIM) / 256, 256, 0, stream>>>(Kf, Kh, cosb, sinb, NKV, 1.0f);
  vtrans<<<(TSEQ * NKV * HDIM) / 256, 256, 0, stream>>>(Vf, Vt);

  // --- causal flash attention (f16 in, f16 ctx out) ---
  attn<<<dim3(TSEQ / 64, NHEADS), dim3(128), 0, stream>>>(Qh, Kh, Vt, ctxh);

  // --- output projection ---
  gemm_h<<<dim3(TSEQ / 128, EMB / 64), blk, 0, stream>>>(ctxh, Woh, out, EMB, EMB);
}
